// Attention_45964740002648
// MI455X (gfx1250) — matrix operand order repaired and checked by reading, not compiled, not run on hardware
//
#include <hip/hip_runtime.h>

#define DIM 2048
#define SEQ 2048
#define NH  32
#define NKV 8
#define HD  64
#define KVD (NKV * HD) /* 512 */

typedef __bf16 bf16;
typedef __bf16 v16bf __attribute__((ext_vector_type(16)));
typedef __bf16 v8bf  __attribute__((ext_vector_type(8)));
typedef float  v8f   __attribute__((ext_vector_type(8)));

__device__ __forceinline__ bf16 f2bf(float f) {
    unsigned u = __builtin_bit_cast(unsigned, f);
    u += 0x7FFFu + ((u >> 16) & 1u);           // round-to-nearest-even
    unsigned short h = (unsigned short)(u >> 16);
    return __builtin_bit_cast(bf16, h);
}

__device__ __forceinline__ v8f v8f_zero() {
    v8f z;
#pragma unroll
    for (int i = 0; i < 8; ++i) z[i] = 0.0f;
    return z;
}

__device__ __forceinline__ v16bf cat16(v8bf lo, v8bf hi) {
    v16bf r;
#pragma unroll
    for (int i = 0; i < 8; ++i) { r[i] = lo[i]; r[i + 8] = hi[i]; }
    return r;
}

// A-matrix 16x32 bf16 fragment (ISA 7.12.2): lane<16 holds row m=lane, k in
// {half*8+0..7} U {16+half*8+0..7}. rowptr points at LDS row start (k=0).
__device__ __forceinline__ v16bf load_frag_a(const bf16* rowptr, int half) {
    v8bf lo = *(const v8bf*)(rowptr + half * 8);
    v8bf hi = *(const v8bf*)(rowptr + 16 + half * 8);
    return cat16(lo, hi);
}

// B-matrix 32x16 bf16 fragment, tile stored [n][k] in LDS: lane n=lane%16,
// k in {half*8+0..7} U {16+half*8+0..7} (same k order as A fragment).
__device__ __forceinline__ v16bf load_frag_b(const bf16* nrowptr, int half) {
    v8bf lo = *(const v8bf*)(nrowptr + half * 8);
    v8bf hi = *(const v8bf*)(nrowptr + 16 + half * 8);
    return cat16(lo, hi);
}

#define WMMA_BF16(a, b, c) \
    __builtin_amdgcn_wmma_f32_16x16x32_bf16(false, (a), false, (b), (short)0, (c), false, false)

// CDNA5 async global->LDS copy (ASYNCcnt-tracked, bypasses VGPRs).
// lds_off: wave-relative LDS byte offset; gaddr: 64-bit global address.
__device__ __forceinline__ void async_g2lds_b128(unsigned lds_off, const void* gp) {
    unsigned long long ga = (unsigned long long)(uintptr_t)gp;
    asm volatile("global_load_async_to_lds_b128 %0, %1, off"
                 :: "v"(lds_off), "v"(ga) : "memory");
}
__device__ __forceinline__ void wait_asynccnt0() {
    asm volatile("s_wait_asynccnt 0x0" ::: "memory");
}
__device__ __forceinline__ unsigned lds_off_of(const void* p) {
    return (unsigned)(uintptr_t)p;  // addrspacecast(LDS->flat) keeps offset in low 32 bits
}

// ---------------------------------------------------------------------------
// C[M,N] = A[M,K] @ B[K,N]; fp32 inputs converted to bf16 while staging to
// LDS; fp32 WMMA accumulation. Block tile 128x128x32, 256 threads (8 waves),
// each wave computes 64x32 (4x2 WMMA tiles).
// ---------------------------------------------------------------------------
template <bool OUT_BF16>
__global__ __launch_bounds__(256) void gemm_xw_bf16(
    const float* __restrict__ A, const float* __restrict__ B,
    void* __restrict__ C, int M, int N, int K) {
    __shared__ bf16 sA[128 * 32];   // [m][k]
    __shared__ bf16 sBt[128 * 32];  // [n][k]  (transposed stage)

    const int tid  = threadIdx.x;
    const int lane = tid & 31, wid = tid >> 5;
    const int half = lane >> 4, lm = lane & 15;
    const int waveM = wid >> 2, waveN = wid & 3;  // 2 x 4 wave grid
    const int m0 = blockIdx.y * 128, n0 = blockIdx.x * 128;

    v8f acc[4][2];
#pragma unroll
    for (int mt = 0; mt < 4; ++mt)
#pragma unroll
        for (int nt = 0; nt < 2; ++nt) acc[mt][nt] = v8f_zero();

    const int aRow = (tid >> 3) * 4;  // 0..124
    const int aCol = (tid & 7) * 4;   // 0..28
    const int bCol = (tid & 31) * 4;  // 0..124
    const int bRow = (tid >> 5) * 4;  // 0..28

    for (int k0 = 0; k0 < K; k0 += 32) {
        if (k0 + 32 < K) {  // L2 prefetch of next k-tile (global_prefetch_b8)
            __builtin_prefetch(A + (size_t)(m0 + aRow) * K + (k0 + 32) + aCol, 0, 1);
            __builtin_prefetch(B + (size_t)(k0 + 32 + bRow) * N + n0 + bCol, 0, 1);
        }
#pragma unroll
        for (int i = 0; i < 4; ++i) {
            const float4 av = *(const float4*)(A + (size_t)(m0 + aRow + i) * K + k0 + aCol);
            bf16* d = &sA[(aRow + i) * 32 + aCol];
            d[0] = f2bf(av.x); d[1] = f2bf(av.y); d[2] = f2bf(av.z); d[3] = f2bf(av.w);
        }
#pragma unroll
        for (int i = 0; i < 4; ++i) {
            const float4 bv = *(const float4*)(B + (size_t)(k0 + bRow + i) * N + n0 + bCol);
            sBt[(bCol + 0) * 32 + bRow + i] = f2bf(bv.x);
            sBt[(bCol + 1) * 32 + bRow + i] = f2bf(bv.y);
            sBt[(bCol + 2) * 32 + bRow + i] = f2bf(bv.z);
            sBt[(bCol + 3) * 32 + bRow + i] = f2bf(bv.w);
        }
        __syncthreads();

        v16bf af[4], bfm[2];
#pragma unroll
        for (int mt = 0; mt < 4; ++mt)
            af[mt] = load_frag_a(&sA[(waveM * 64 + mt * 16 + lm) * 32], half);
#pragma unroll
        for (int nt = 0; nt < 2; ++nt)
            bfm[nt] = load_frag_b(&sBt[(waveN * 32 + nt * 16 + lm) * 32], half);
#pragma unroll
        for (int mt = 0; mt < 4; ++mt)
#pragma unroll
            for (int nt = 0; nt < 2; ++nt)
                acc[mt][nt] = WMMA_BF16(af[mt], bfm[nt], acc[mt][nt]);
        __syncthreads();
    }

#pragma unroll
    for (int mt = 0; mt < 4; ++mt)
#pragma unroll
        for (int nt = 0; nt < 2; ++nt) {
            const int gcol  = n0 + waveN * 32 + nt * 16 + lm;
            const int grow0 = m0 + waveM * 64 + mt * 16 + half * 8;
#pragma unroll
            for (int r = 0; r < 8; ++r) {
                const float v = acc[mt][nt][r];
                if constexpr (OUT_BF16)
                    ((bf16*)C)[(size_t)(grow0 + r) * N + gcol] = f2bf(v);
                else
                    ((float*)C)[(size_t)(grow0 + r) * N + gcol] = v;
            }
        }
}

// ---------------------------------------------------------------------------
// RoPE: interleaved pairs (2p, 2p+1) within each head; fp32 in, bf16 out.
// ---------------------------------------------------------------------------
__global__ __launch_bounds__(256) void rope_bf16(
    const float* __restrict__ in, const float* __restrict__ cs,
    const float* __restrict__ sn, bf16* __restrict__ out, int nheads) {
    const int idx   = blockIdx.x * 256 + threadIdx.x;
    const int total = SEQ * nheads * (HD / 2);
    if (idx >= total) return;
    const int p = idx & 31;
    const int h = (idx >> 5) % nheads;
    const int s = idx / (nheads * 32);
    const float c = cs[s * 32 + p], si = sn[s * 32 + p];
    const size_t base = ((size_t)s * nheads + h) * HD + 2 * p;
    const float xr = in[base], xi = in[base + 1];
    out[base]     = f2bf(xr * c - xi * si);
    out[base + 1] = f2bf(xr * si + xi * c);
}

// ---------------------------------------------------------------------------
// Flash attention, causal, GQA (kv head = h/4). One block per (64-query tile,
// head). 256 threads = 8 waves in a 4(M) x 2(N-of-64) grid. Online softmax in
// fp32; both GEMMs (QK^T, PV) via bf16 WMMA. Q/K tiles staged with CDNA5
// async global->LDS DMA (no VGPR round-trip); V staged via VGPRs (needs
// transpose into [d][key]).
// ---------------------------------------------------------------------------
__global__ __launch_bounds__(256) void flash_attn(
    const bf16* __restrict__ Q, const bf16* __restrict__ K,
    const bf16* __restrict__ V, float* __restrict__ O) {
    __shared__ bf16  sQ[64 * 64];   // [q][d]
    __shared__ bf16  sK[64 * 64];   // [key][d]  == B[n][k] layout for QK^T
    __shared__ bf16  sVt[64 * 64];  // [d][key]  == B[n][k] layout for P@V
    __shared__ float sS[64 * 64];   // masked, scaled scores
    __shared__ bf16  sP[64 * 64];   // exp probabilities
    __shared__ float sMax[64], sSum[64], sFac[64];

    const int tid  = threadIdx.x;
    const int lane = tid & 31, wid = tid >> 5;
    const int half = lane >> 4, lm = lane & 15;
    const int wm = wid >> 1, wn = wid & 1;
    const int h = blockIdx.y;
    const int kvh = h >> 2;  // N_REP = 4
    const int q0 = blockIdx.x * 64;

    // async stage Q tile: 64 rows x 128B, 512 x b128 segments, 2 per thread
#pragma unroll
    for (int it = 0; it < 2; ++it) {
        const int sidx = tid + it * 256;
        const int row = sidx >> 3, part = sidx & 7;   // 8 x 16B per row
        async_g2lds_b128(lds_off_of(&sQ[row * 64 + part * 8]),
                         Q + (size_t)(q0 + row) * DIM + h * HD + part * 8);
    }
    if (tid < 64) { sMax[tid] = -1e30f; sSum[tid] = 0.0f; }

    v8f oacc[2];
    oacc[0] = v8f_zero();
    oacc[1] = v8f_zero();

    const int nkt = blockIdx.x + 1;  // causal: key tiles 0..query tile
    for (int kt = 0; kt < nkt; ++kt) {
        {   // async stage K row-major; V via VGPRs, transposed into sVt
#pragma unroll
            for (int it = 0; it < 2; ++it) {
                const int sidx = tid + it * 256;
                const int row = sidx >> 3, part = sidx & 7;
                async_g2lds_b128(lds_off_of(&sK[row * 64 + part * 8]),
                                 K + (size_t)(kt * 64 + row) * KVD + kvh * HD + part * 8);
            }
            const int row = tid >> 2, chunk = tid & 3;
            const bf16* gv = V + (size_t)(kt * 64 + row) * KVD + kvh * HD + chunk * 16;
            v8bf v0 = *(const v8bf*)gv;
            v8bf v1 = *(const v8bf*)(gv + 8);
#pragma unroll
            for (int e = 0; e < 8; ++e) {
                sVt[(chunk * 16 + e) * 64 + row]     = v0[e];
                sVt[(chunk * 16 + 8 + e) * 64 + row] = v1[e];
            }
        }
        wait_asynccnt0();   // Q (first iter) + K DMA complete before use
        __syncthreads();

        // S = scale * (Q K^T), causal-masked. Wave: rows wm*16.., keys wn*32..
        v8f sacc[2];
        sacc[0] = v8f_zero();
        sacc[1] = v8f_zero();
        const v16bf aq0 = load_frag_a(&sQ[(wm * 16 + lm) * 64 + 0], half);
        const v16bf aq1 = load_frag_a(&sQ[(wm * 16 + lm) * 64 + 32], half);
#pragma unroll
        for (int nt = 0; nt < 2; ++nt) {
            const v16bf b0 = load_frag_b(&sK[(wn * 32 + nt * 16 + lm) * 64 + 0], half);
            const v16bf b1 = load_frag_b(&sK[(wn * 32 + nt * 16 + lm) * 64 + 32], half);
            sacc[nt] = WMMA_BF16(aq0, b0, sacc[nt]);
            sacc[nt] = WMMA_BF16(aq1, b1, sacc[nt]);
        }
#pragma unroll
        for (int nt = 0; nt < 2; ++nt) {
            const int kcol  = wn * 32 + nt * 16 + lm;
            const int kglob = kt * 64 + kcol;
#pragma unroll
            for (int r = 0; r < 8; ++r) {
                const int ml = wm * 16 + half * 8 + r;
                float v = sacc[nt][r] * 0.125f;  // 1/sqrt(64)
                if (kglob > q0 + ml) v = -1e30f;
                sS[ml * 64 + kcol] = v;
            }
        }
        __syncthreads();

        // online softmax, one row per thread (fp32)
        if (tid < 64) {
            const int m = tid;
            const float mOld = sMax[m];
            float rmax = mOld;
            for (int j = 0; j < 64; ++j) rmax = fmaxf(rmax, sS[m * 64 + j]);
            const float fac = __expf(mOld - rmax);
            float sum = 0.0f;
            for (int j = 0; j < 64; ++j) {
                const float p = __expf(sS[m * 64 + j] - rmax);
                sP[m * 64 + j] = f2bf(p);
                sum += p;
            }
            sSum[m] = sSum[m] * fac + sum;
            sMax[m] = rmax;
            sFac[m] = fac;
        }
        __syncthreads();

        // rescale running output, then O += P @ V
        {
            float facr[8];
#pragma unroll
            for (int r = 0; r < 8; ++r) facr[r] = sFac[wm * 16 + half * 8 + r];
#pragma unroll
            for (int nt = 0; nt < 2; ++nt)
#pragma unroll
                for (int r = 0; r < 8; ++r) oacc[nt][r] *= facr[r];

            const v16bf ap0 = load_frag_a(&sP[(wm * 16 + lm) * 64 + 0], half);
            const v16bf ap1 = load_frag_a(&sP[(wm * 16 + lm) * 64 + 32], half);
#pragma unroll
            for (int nt = 0; nt < 2; ++nt) {
                const v16bf b0 = load_frag_b(&sVt[(wn * 32 + nt * 16 + lm) * 64 + 0], half);
                const v16bf b1 = load_frag_b(&sVt[(wn * 32 + nt * 16 + lm) * 64 + 32], half);
                oacc[nt] = WMMA_BF16(ap0, b0, oacc[nt]);
                oacc[nt] = WMMA_BF16(ap1, b1, oacc[nt]);
            }
        }
        __syncthreads();
    }

    // normalize and store fp32
#pragma unroll
    for (int nt = 0; nt < 2; ++nt)
#pragma unroll
        for (int r = 0; r < 8; ++r) {
            const int ml = wm * 16 + half * 8 + r;
            const float inv = 1.0f / sSum[ml];
            O[(size_t)(q0 + ml) * DIM + h * HD + wn * 32 + nt * 16 + lm] = oacc[nt][r] * inv;
        }
}

// ---------------------------------------------------------------------------
extern "C" void kernel_launch(void* const* d_in, const int* in_sizes, int n_in,
                              void* d_out, int out_size, void* d_ws, size_t ws_size,
                              hipStream_t stream) {
    (void)in_sizes; (void)n_in; (void)out_size; (void)ws_size;
    const float* x  = (const float*)d_in[0];
    const float* fc = (const float*)d_in[1];
    const float* fs = (const float*)d_in[2];
    const float* wq = (const float*)d_in[3];
    const float* wk = (const float*)d_in[4];
    const float* wv = (const float*)d_in[5];
    const float* wo = (const float*)d_in[6];

    char* w = (char*)d_ws;
    float* Qf = (float*)(w);                               // 16 MB (pre-rope Q, fp32)
    float* Kf = (float*)(w + (size_t)16 * 1024 * 1024);    //  4 MB (pre-rope K, fp32)
    bf16*  Qb = (bf16*)(w + (size_t)20 * 1024 * 1024);     //  8 MB
    bf16*  Kb = (bf16*)(w + (size_t)28 * 1024 * 1024);     //  2 MB
    bf16*  Vb = (bf16*)(w + (size_t)30 * 1024 * 1024);     //  2 MB
    float* Of = (float*)(w + (size_t)32 * 1024 * 1024);    // 16 MB (attn out, fp32)

    const dim3 blk(256, 1, 1);

    gemm_xw_bf16<false><<<dim3(DIM / 128, SEQ / 128), blk, 0, stream>>>(x, wq, Qf, SEQ, DIM, DIM);
    gemm_xw_bf16<false><<<dim3(KVD / 128, SEQ / 128), blk, 0, stream>>>(x, wk, Kf, SEQ, KVD, DIM);
    gemm_xw_bf16<true ><<<dim3(KVD / 128, SEQ / 128), blk, 0, stream>>>(x, wv, Vb, SEQ, KVD, DIM);

    rope_bf16<<<dim3((SEQ * NH  * (HD / 2)) / 256), blk, 0, stream>>>(Qf, fc, fs, Qb, NH);
    rope_bf16<<<dim3((SEQ * NKV * (HD / 2)) / 256), blk, 0, stream>>>(Kf, fc, fs, Kb, NKV);

    flash_attn<<<dim3(SEQ / 64, NH), blk, 0, stream>>>(Qb, Kb, Vb, Of);

    gemm_xw_bf16<false><<<dim3(DIM / 128, SEQ / 128), blk, 0, stream>>>(Of, wo, (float*)d_out, SEQ, DIM, DIM);
}
